// Attention_824633721466
// MI455X (gfx1250) — matrix operand order repaired and checked by reading, not compiled, not run on hardware
//
#include <hip/hip_runtime.h>
#include <math.h>

// ---------------------------------------------------------------------------
// Problem constants (fixed by the reference)
// ---------------------------------------------------------------------------
#define BB   8          // batch
#define NN   1024       // sequence
#define DIMM 256        // model dim == INNER
#define HH   8          // heads
#define DHH  32         // head dim
#define TIE  4
#define NBB  (BB / TIE) // tied batch groups = 2
#define QSCALE 0.17677669529663687f   // DH^-0.5

typedef _Float16 f16;
typedef __attribute__((ext_vector_type(16))) _Float16 v16h;
typedef __attribute__((ext_vector_type(8)))  _Float16 v8h;
typedef __attribute__((ext_vector_type(8)))  float    v8f;

// ---------------------------------------------------------------------------
// WMMA fragment loaders (16x16x32 f16; this code's contiguous k order, same for A and B)
//
// A (16x32, MxK): lane L holds row M = L%16.
//   lanes 0-15 : halves 0-15 -> K 0..15
//   lanes 16-31: halves 0-15 -> K 16..31
// B (32x16, KxN): lane L holds col N = L%16.
//   lanes 0-15 : halves 0-15 -> K 0..15
//   lanes 16-31: halves 0-15 -> K 16..31
// C/D (16x16 f32): vgpr g, lanes 0-15 -> M=g, lanes 16-31 -> M=g+8; N=L%16.
// ---------------------------------------------------------------------------
static __device__ __forceinline__ v16h cat8(v8h lo, v8h hi) {
  v16h r;
#pragma unroll
  for (int i = 0; i < 8; ++i) { r[i] = lo[i]; r[i + 8] = hi[i]; }
  return r;
}
static __device__ __forceinline__ v16h load_afrag(const f16* row, int halfw) {
  const v8h* p0 = (const v8h*)(row + halfw * 16);
  const v8h* p1 = (const v8h*)(row + halfw * 16 + 8);
  return cat8(*p0, *p1);
}
static __device__ __forceinline__ v16h load_bfrag(const f16* col, int halfw) {
  const v8h* p = (const v8h*)(col + halfw * 16);
  return cat8(p[0], p[1]);
}
static __device__ __forceinline__ v8f wmma_f16(v16h a, v16h b, v8f c) {
  return __builtin_amdgcn_wmma_f32_16x16x32_f16(false, a, false, b,
                                                (short)0, c, false, false);
}

// ---------------------------------------------------------------------------
// CDNA5 async global->LDS copy (ASYNCcnt path, CDNA5 ISA)
// LDS address = low 32 bits of the flat pointer (allocation-relative).
// ---------------------------------------------------------------------------
static __device__ __forceinline__ void async_g2l_b128(f16* lds_ptr,
                                                      const f16* gptr) {
  unsigned l = (unsigned)(size_t)lds_ptr;
  unsigned long long g = (unsigned long long)(size_t)gptr;
  asm volatile("global_load_async_to_lds_b128 %0, %1, off"
               :: "v"(l), "v"(g) : "memory");
}
static __device__ __forceinline__ void async_wait0() {
  asm volatile("s_wait_asynccnt 0" ::: "memory");
}

// ---------------------------------------------------------------------------
// Prep kernels: f32 -> f16 conversions / transposes / tie-mean
// ---------------------------------------------------------------------------
__global__ void prep_weights(const float* __restrict__ Wq,
                             const float* __restrict__ Wkv,
                             const float* __restrict__ Wg,
                             const float* __restrict__ Wout,
                             f16* __restrict__ WqT, f16* __restrict__ WkT,
                             f16* __restrict__ WvT, f16* __restrict__ WgT,
                             f16* __restrict__ WoT) {
  int idx = blockIdx.x * blockDim.x + threadIdx.x;
  if (idx >= 5 * DIMM * DIMM) return;
  int w = idx / (DIMM * DIMM);
  int rem = idx % (DIMM * DIMM);
  int c = rem / DIMM;   // output column
  int i = rem % DIMM;   // input (contraction) index
  float v;
  f16* dst;
  switch (w) {
    case 0: v = Wq[i * DIMM + c];             dst = WqT; break;
    case 1: v = Wkv[i * 2 * DIMM + c];        dst = WkT; break;
    case 2: v = Wkv[i * 2 * DIMM + DIMM + c]; dst = WvT; break;
    case 3: v = Wg[i * DIMM + c];             dst = WgT; break;
    default: v = Wout[i * DIMM + c];          dst = WoT; break;
  }
  dst[c * DIMM + i] = (f16)v;   // [outcol][in] : K-contiguous per column
}

__global__ void prep_x(const float* __restrict__ x, f16* __restrict__ xh) {
  int idx = blockIdx.x * blockDim.x + threadIdx.x;
  if (idx >= BB * NN * DIMM) return;
  xh[idx] = (f16)x[idx];
}

__global__ void prep_xm(const float* __restrict__ x, f16* __restrict__ xmh) {
  int idx = blockIdx.x * blockDim.x + threadIdx.x;
  if (idx >= NBB * NN * DIMM) return;
  int bb = idx / (NN * DIMM);
  int rem = idx % (NN * DIMM);
  float s = 0.f;
#pragma unroll
  for (int t = 0; t < TIE; ++t)
    s += x[(size_t)(bb * TIE + t) * NN * DIMM + rem];
  xmh[idx] = (f16)(s * 0.25f);
}

// ---------------------------------------------------------------------------
// WMMA GEMM:  C = A(Mx256) * W(256x256).  One wave computes a 16x64 strip
// (4 N-tiles share one A fragment -> 4x less A traffic, 32 static WMMAs).
//   mode 0: Qm (scale, [bb,h,n,dh])   mode 1: K ([b,h,n,dh])
//   mode 2: V^T ([b,h,dh,n])          mode 3: gates (sigmoid(v+bg))
//   mode 4: OUT (v+bout -> f32 d_out)
// ---------------------------------------------------------------------------
__global__ void gemm16(const f16* __restrict__ A, const f16* __restrict__ WT,
                       const float* __restrict__ bias, f16* __restrict__ outh,
                       float* __restrict__ outf, int M, int mode) {
  int lane = threadIdx.x & 31;
  int wv   = threadIdx.x >> 5;
  int wave = blockIdx.x * 4 + wv;    // (M/16)*4 waves total
  int tm = wave >> 2;                // row tile
  int tg = wave & 3;                 // 64-column group
  if (tm * 16 >= M) return;
  int halfw = lane >> 4;
  int ln    = lane & 15;

  const f16* arow = A  + (size_t)(tm * 16 + ln) * DIMM;
  const f16* wcol = WT + (size_t)(tg * 64 + ln) * DIMM;

  v8f acc[4] = {{}, {}, {}, {}};
#pragma unroll
  for (int kk = 0; kk < DIMM; kk += 32) {
    v16h a = load_afrag(arow + kk, halfw);
#pragma unroll
    for (int s = 0; s < 4; ++s)
      acc[s] = wmma_f16(a, load_bfrag(wcol + (size_t)s * 16 * DIMM + kk, halfw),
                        acc[s]);
  }

#pragma unroll
  for (int s = 0; s < 4; ++s) {
#pragma unroll
    for (int g = 0; g < 8; ++g) {
      int r = tm * 16 + g + halfw * 8;     // global row (b*N + n)
      int c = tg * 64 + s * 16 + ln;       // global col (h*DH + d)
      float v = acc[s][g];
      int b = r >> 10, n = r & 1023;
      int h = c >> 5,  d = c & 31;
      switch (mode) {
        case 0:
          outh[(((size_t)b * HH + h) * NN + n) * DHH + d] = (f16)(v * QSCALE);
          break;
        case 1:
          outh[(((size_t)b * HH + h) * NN + n) * DHH + d] = (f16)v;
          break;
        case 2:
          outh[(((size_t)b * HH + h) * DHH + d) * NN + n] = (f16)v;
          break;
        case 3:
          outh[(size_t)r * DIMM + c] =
              (f16)(1.0f / (1.0f + __expf(-(v + bias[c]))));
          break;
        default:
          outf[(size_t)r * DIMM + c] = v + bias[c];
          break;
      }
    }
  }
}

// ---------------------------------------------------------------------------
// Flash attention: 4 waves/block share one (b,h); K/V 32-wide chunks are
// staged block-wide into LDS via global_load_async_to_lds_b128 (ASYNCcnt),
// fragments then come from ds_load_b128. Online softmax, all matmuls WMMA.
// ---------------------------------------------------------------------------
__global__ void __launch_bounds__(128)
flash_attn(const f16* __restrict__ Qm, const f16* __restrict__ Kh,
           const f16* __restrict__ Vt, const f16* __restrict__ Gh,
           const float* __restrict__ bias, const unsigned char* __restrict__ mask,
           f16* __restrict__ Oh) {
  __shared__ __align__(16) f16 klds[32 * 32];      // [j'][d]  2KB
  __shared__ __align__(16) f16 vlds[32 * 32];      // [d][j']  2KB
  __shared__ __align__(16) f16 plds[4][16 * 32];   // per-wave P tile

  int tid  = threadIdx.x;
  int lane = tid & 31;
  int wv   = tid >> 5;
  int halfw = lane >> 4;
  int ln    = lane & 15;

  int tile = blockIdx.x * 4 + wv;       // 0 .. 4095
  int it = tile & 63;                   // row tile (4 consecutive per block)
  int bh = tile >> 6;                   // same (b,h) for the whole block
  int h  = bh & 7;
  int b  = bh >> 3;
  int bb = b >> 2;                      // tie group
  int i0 = it * 16;

  const f16* kbase = Kh + ((size_t)b * HH + h) * NN * DHH;
  const f16* vbase = Vt + ((size_t)b * HH + h) * DHH * NN;
  const float* brow = bias + ((size_t)h * NN + i0) * NN;

  // Q fragment (A layout): row m = ln, K = d (0..31)
  const f16* qrow = Qm + (((size_t)bb * HH + h) * NN + i0 + ln) * DHH;
  v16h qA = load_afrag(qrow, halfw);

  float rmax[8], rsum[8];
  unsigned char mrow[8];
#pragma unroll
  for (int g = 0; g < 8; ++g) {
    rmax[g] = -3.0e38f;
    rsum[g] = 0.0f;
    mrow[g] = mask[b * NN + i0 + g + halfw * 8];
  }
  v8f o0 = {}, o1 = {};

  // thread's share of the block-wide K/V staging (1 b128 each)
  int srow = tid >> 2;                  // 0..31
  int sseg = (tid & 3) * 8;             // half offset within row
  f16* kdst = klds + tid * 8;           // row-major [j'][d]
  f16* vdst = vlds + tid * 8;           // row-major [d][j']

  // stage chunk j0 = 0
  async_g2l_b128(kdst, kbase + (size_t)srow * DHH + sseg);
  async_g2l_b128(vdst, vbase + (size_t)srow * NN + sseg);

  for (int j0 = 0; j0 < NN; j0 += 32) {
    __builtin_prefetch(brow + j0 + 32, 0, 3);   // bias = dominant stream

    async_wait0();
    __syncthreads();                    // K/V chunk visible to all waves

    // fragments from LDS (ds_load_b128 x2 each)
    v16h kf0 = load_bfrag(&klds[(0 + ln) * 32], halfw);
    v16h kf1 = load_bfrag(&klds[(16 + ln) * 32], halfw);
    v16h vf0 = load_bfrag(&vlds[(0 + ln) * 32], halfw);
    v16h vf1 = load_bfrag(&vlds[(16 + ln) * 32], halfw);

    v8f zero = {};
    v8f d0 = wmma_f16(qA, kf0, zero);
    v8f d1 = wmma_f16(qA, kf1, zero);

    unsigned char mk0 = mask[b * NN + j0 + ln];
    unsigned char mk1 = mask[b * NN + j0 + 16 + ln];
#pragma unroll
    for (int g = 0; g < 8; ++g) {
      const float* bp = brow + (size_t)(g + halfw * 8) * NN + j0;
      d0[g] = (mrow[g] && mk0) ? d0[g] + bp[ln]      : -3.0e38f;
      d1[g] = (mrow[g] && mk1) ? d1[g] + bp[16 + ln] : -3.0e38f;

      float t = fmaxf(d0[g], d1[g]);
#pragma unroll
      for (int m = 1; m < 16; m <<= 1) t = fmaxf(t, __shfl_xor(t, m));
      float nm = fmaxf(rmax[g], t);
      float corr = __expf(rmax[g] - nm);
      rmax[g] = nm;
      float p0 = __expf(d0[g] - nm);
      float p1 = __expf(d1[g] - nm);
      float s = p0 + p1;
#pragma unroll
      for (int m = 1; m < 16; m <<= 1) s += __shfl_xor(s, m);
      rsum[g] = rsum[g] * corr + s;
      o0[g] *= corr;
      o1[g] *= corr;
      plds[wv][(g + halfw * 8) * 32 + ln]      = (f16)p0;
      plds[wv][(g + halfw * 8) * 32 + 16 + ln] = (f16)p1;
    }
    __syncthreads();   // P visible; K/V LDS fully consumed (vf in regs)

    // prefill next chunk while we do the PV WMMAs
    if (j0 + 32 < NN) {
      async_g2l_b128(kdst, kbase + (size_t)(j0 + 32 + srow) * DHH + sseg);
      async_g2l_b128(vdst, vbase + (size_t)srow * NN + j0 + 32 + sseg);
    }

    v16h pA = load_afrag(&plds[wv][ln * 32], halfw);
    o0 = wmma_f16(pA, vf0, o0);
    o1 = wmma_f16(pA, vf1, o1);
  }

  // normalize, gate, store f16 [b, n, h*DH + d]
#pragma unroll
  for (int g = 0; g < 8; ++g) {
    float inv = 1.0f / fmaxf(rsum[g], 1e-20f);
    int i = i0 + g + halfw * 8;
    size_t base = ((size_t)b * NN + i) * DIMM + h * DHH;
    float g0 = (float)Gh[base + ln];
    float g1 = (float)Gh[base + 16 + ln];
    Oh[base + ln]      = (f16)(o0[g] * inv * g0);
    Oh[base + 16 + ln] = (f16)(o1[g] * inv * g1);
  }
}

// ---------------------------------------------------------------------------
// Host launcher
// ---------------------------------------------------------------------------
extern "C" void kernel_launch(void* const* d_in, const int* in_sizes, int n_in,
                              void* d_out, int out_size, void* d_ws, size_t ws_size,
                              hipStream_t stream) {
  const float* x    = (const float*)d_in[0];
  const unsigned char* mask = (const unsigned char*)d_in[1];  // jnp bool = 1B
  const float* bias = (const float*)d_in[2];
  // d_in[3] = tie_dim (fixed 4, baked in)
  const float* Wq   = (const float*)d_in[4];
  const float* Wkv  = (const float*)d_in[5];
  const float* Wout = (const float*)d_in[6];
  const float* bout = (const float*)d_in[7];
  const float* Wg   = (const float*)d_in[8];
  const float* bg   = (const float*)d_in[9];
  float* out = (float*)d_out;

  // workspace carve-up (f16 elements), ~23.7 MB total
  f16* p = (f16*)d_ws;
  f16* x_h  = p; p += (size_t)BB * NN * DIMM;
  f16* xm_h = p; p += (size_t)NBB * NN * DIMM;
  f16* WqT  = p; p += DIMM * DIMM;
  f16* WkT  = p; p += DIMM * DIMM;
  f16* WvT  = p; p += DIMM * DIMM;
  f16* WgT  = p; p += DIMM * DIMM;
  f16* WoT  = p; p += DIMM * DIMM;
  f16* Qm   = p; p += (size_t)NBB * HH * NN * DHH;
  f16* Kh   = p; p += (size_t)BB * HH * NN * DHH;
  f16* Vt   = p; p += (size_t)BB * HH * NN * DHH;
  f16* Gh   = p; p += (size_t)BB * NN * DIMM;
  f16* Oh   = p; p += (size_t)BB * NN * DIMM;

  // 1) prep
  prep_weights<<<(5 * DIMM * DIMM + 255) / 256, 256, 0, stream>>>(
      Wq, Wkv, Wg, Wout, WqT, WkT, WvT, WgT, WoT);
  prep_x<<<(BB * NN * DIMM + 255) / 256, 256, 0, stream>>>(x, x_h);
  prep_xm<<<(NBB * NN * DIMM + 255) / 256, 256, 0, stream>>>(x, xm_h);

  // 2) projections (WMMA GEMMs): blocks = M/16 (4 waves x 16x64 strip each)
  int Mfull = BB * NN;      // 8192
  int Mtied = NBB * NN;     // 2048
  gemm16<<<Mtied / 16, 128, 0, stream>>>(xm_h, WqT, nullptr, Qm, nullptr, Mtied, 0);
  gemm16<<<Mfull / 16, 128, 0, stream>>>(x_h,  WkT, nullptr, Kh, nullptr, Mfull, 1);
  gemm16<<<Mfull / 16, 128, 0, stream>>>(x_h,  WvT, nullptr, Vt, nullptr, Mfull, 2);
  gemm16<<<Mfull / 16, 128, 0, stream>>>(x_h,  WgT, bg,      Gh, nullptr, Mfull, 3);

  // 3) flash attention + gating (async K/V staging into LDS)
  flash_attn<<<(BB * HH * (NN / 16)) / 4, 128, 0, stream>>>(
      Qm, Kh, Vt, Gh, bias, mask, Oh);

  // 4) output projection -> f32 d_out
  gemm16<<<Mfull / 16, 128, 0, stream>>>(Oh, WoT, bout, nullptr, out, Mfull, 4);
}
